// layer_30099130810406
// MI455X (gfx1250) — matrix operand order repaired and checked by reading, not compiled, not run on hardware
//
#include <hip/hip_runtime.h>

// ---------------------------------------------------------------------------
// MI455X (gfx1250, wave32) transformer layer:
//   rmsnorm -> QKV (bf16 WMMA) -> RoPE -> sliding-window flash attention
//   (bf16 WMMA, transposed-score formulation) -> Wo + residual -> rmsnorm
//   -> router top-2 -> dense-expert MoE (bf16 WMMA, SiLU) -> out
// All matmuls use V_WMMA_F32_16X16X32_BF16 with fp32 accumulation.
// GEMM B-tiles are staged in LDS via GLOBAL_LOAD_ASYNC_TO_LDS_B128
// (ASYNCcnt-tracked, double-buffered) so one block's 8 waves share each tile.
// Attention: wave-uniform values go through readfirstlane so tile loops and
// the edge/interior softmax split lower to scalar branches, and the P->B
// fragment conversion shuffles packed bf16x2 dwords (16 shfl, not 32).
// ---------------------------------------------------------------------------

typedef __attribute__((ext_vector_type(16))) __bf16   v16bf;
typedef __attribute__((ext_vector_type(8)))  float    v8f;
typedef __attribute__((ext_vector_type(4)))  unsigned v4u;

constexpr int Bc = 2, Sc = 2048, Dc = 1024, Hc = 16, HKc = 4, HDc = 64, WINc = 1024;
constexpr int Ec = 8, DHc = 2048;
constexpr int Tc = Bc * Sc;              // 4096 tokens

union AB16 { unsigned u[8]; v4u q[2]; __bf16 b[16]; v16bf v; };

static __device__ __forceinline__ v8f v8zero() {
  v8f z = {0.f, 0.f, 0.f, 0.f, 0.f, 0.f, 0.f, 0.f};
  return z;
}

static __device__ __forceinline__ unsigned pack_bf16x2(float a, float b) {
  union { __bf16 h[2]; unsigned u; } t;   // fuses to v_cvt_pk_bf16_f32
  t.h[0] = (__bf16)a; t.h[1] = (__bf16)b;
  return t.u;
}

// ---------------------------------------------------------------------------
// fp32 [K,N] -> bf16 [N,K] (transpose so GEMM B-side loads are contiguous in K)
// ---------------------------------------------------------------------------
__global__ void convt_kernel(const float* __restrict__ in, __bf16* __restrict__ outp,
                             int K, int N) {
  int idx = blockIdx.x * 256 + threadIdx.x;  // flat over [K,N]
  int k = idx / N, n = idx - k * N;
  outp[(size_t)n * K + k] = (__bf16)in[idx];
}

// ---------------------------------------------------------------------------
// RMSNorm: fp32 [T,D] -> bf16 [T,D]; one block per token
// ---------------------------------------------------------------------------
__global__ __launch_bounds__(256) void rmsnorm_kernel(const float* __restrict__ x,
    const float* __restrict__ w, __bf16* __restrict__ outp, float eps) {
  __shared__ float red[8];
  const int token = blockIdx.x;
  const float* xr = x + (size_t)token * Dc;
  float ss = 0.f;
  for (int i = threadIdx.x; i < Dc; i += 256) { float v = xr[i]; ss += v * v; }
#pragma unroll
  for (int o = 16; o; o >>= 1) ss += __shfl_xor(ss, o, 32);
  if ((threadIdx.x & 31) == 0) red[threadIdx.x >> 5] = ss;
  __syncthreads();
  float tot = 0.f;
#pragma unroll
  for (int i = 0; i < 8; ++i) tot += red[i];
  const float scl = rsqrtf(tot * (1.f / Dc) + eps);
  __bf16* orow = outp + (size_t)token * Dc;
  for (int i = threadIdx.x; i < Dc; i += 256) orow[i] = (__bf16)(xr[i] * scl * w[i]);
}

// ---------------------------------------------------------------------------
// bf16 WMMA GEMM with async-LDS B staging.
//   A: bf16 [M,Kd] row-major.  BT: bf16 [N,Kd] (B^T, contiguous in K).
// Block = 8 waves; wave w owns rows [bx*128 + w*16, +16), block owns cols
// [by*64, +64).  Per 32-K step the block DMAs the shared 64x32 B-tile
// (4 KB) into LDS with GLOBAL_LOAD_ASYNC_TO_LDS_B128 (one 16B chunk per
// thread), double-buffered: next tile's async copy is issued before waiting
// (s_wait_asynccnt 0x1) on the current one, overlapping DMA with WMMAs.
// A-fragments stream straight from global (per-wave rows, no sharing).
// EPI: 0 = Cf = acc
//      1 = h = res + acc; Cf = h; Cf2 = h            (Wo + residual, dual store)
//      2 = Cb = bf16(silu(acc))                      (MoE up-proj)
//      3 = Cf += gate[row*gstride] * acc             (MoE down-proj accumulate)
// ---------------------------------------------------------------------------
template <int EPI>
__global__ __launch_bounds__(256) void gemm_bf16(
    const __bf16* __restrict__ A, const __bf16* __restrict__ BT,
    float* __restrict__ Cf, float* __restrict__ Cf2, __bf16* __restrict__ Cb,
    const float* __restrict__ res, const float* __restrict__ gate, int gstride,
    int M, int N, int Kd) {
  __shared__ __bf16 ldsB[2][64 * 32];    // two 4 KB B-tile buffers

  const int tid  = threadIdx.x;
  const int wave = __builtin_amdgcn_readfirstlane(tid >> 5);   // provably uniform
  const int lane = tid & 31;
  const int half = lane >> 4, ln = lane & 15;
  const int row0 = (blockIdx.x * 8 + wave) * 16;
  const int col0 = blockIdx.y * 64;

  // per-thread 16B slice of the 64x32 B-tile: col bn, K-offset bo
  const int bn = tid >> 2, bo = (tid & 3) * 8;

  v8f acc[4];
#pragma unroll
  for (int t = 0; t < 4; ++t) acc[t] = v8zero();

  const __bf16* arow = A + (size_t)(row0 + ln) * Kd;

  // prologue: DMA first B-tile
  {
    unsigned lo = (unsigned)(uintptr_t)&ldsB[0][bn * 32 + bo];
    const __bf16* gp = BT + (size_t)(col0 + bn) * Kd + bo;
    asm volatile("global_load_async_to_lds_b128 %0, %1, off"
                 :: "v"(lo), "v"(gp) : "memory");
  }

  for (int k0 = 0; k0 < Kd; k0 += 32) {
    const int cur = (k0 >> 5) & 1;
    const bool more = (k0 + 32) < Kd;
    if (more) {                                        // prefetch next tile
      unsigned lo = (unsigned)(uintptr_t)&ldsB[cur ^ 1][bn * 32 + bo];
      const __bf16* gp = BT + (size_t)(col0 + bn) * Kd + (k0 + 32) + bo;
      asm volatile("global_load_async_to_lds_b128 %0, %1, off"
                   :: "v"(lo), "v"(gp) : "memory");
    }

    // A-fragment (16-bit A 16x32 layout), overlaps with async wait
    AB16 af;
#pragma unroll
    for (int v2 = 0; v2 < 4; ++v2) {
      af.u[v2]     = *(const unsigned*)(arow + k0 +      half * 8 + 2 * v2);
      af.u[4 + v2] = *(const unsigned*)(arow + k0 + 16 + half * 8 + 2 * v2);
    }

    if (more) asm volatile("s_wait_asynccnt 0x1" ::: "memory");  // cur done, prefetch in flight
    else      asm volatile("s_wait_asynccnt 0x0" ::: "memory");  // drain on last step
    __syncthreads();                                   // all waves' DMAs visible

#pragma unroll
    for (int t = 0; t < 4; ++t) {
      AB16 bf_;                                        // B-frag (16-bit B 32x16 layout):
      const __bf16* bp = &ldsB[cur][(t * 16 + ln) * 32 + half * 8];
      bf_.q[0] = *(const v4u*)bp;                      // K [8h,8h+8) and [16+8h,+8) -> 2x ds_load_b128
      bf_.q[1] = *(const v4u*)(bp + 16);
      acc[t] = __builtin_amdgcn_wmma_f32_16x16x32_bf16(
          false, af.v, false, bf_.v, (short)0, acc[t], false, false);
    }
    __syncthreads();                                   // tile fully consumed before overwrite
  }

#pragma unroll
  for (int t = 0; t < 4; ++t) {
#pragma unroll
    for (int v2 = 0; v2 < 8; ++v2) {                   // C layout: M = v2 + half*8
      const int r = row0 + v2 + half * 8;
      const int c = col0 + t * 16 + ln;
      const size_t idx = (size_t)r * N + c;
      const float a = acc[t][v2];
      if (EPI == 0) {
        Cf[idx] = a;
      } else if (EPI == 1) {
        const float hv = res[idx] + a;
        Cf[idx] = hv; Cf2[idx] = hv;
      } else if (EPI == 2) {
        Cb[idx] = (__bf16)(a / (1.f + __expf(-a)));
      } else {
        Cf[idx] += gate[(size_t)r * gstride] * a;
      }
    }
  }
}

// ---------------------------------------------------------------------------
// RoPE + layout change: fp32 [B,S,nh,HD] -> bf16 [B,nh,S,HD] (optionally scaled)
// one thread per (b,s,h,pair)
// ---------------------------------------------------------------------------
__global__ void rope_kernel(const float* __restrict__ in, const float* __restrict__ cosb,
                            const float* __restrict__ sinb, __bf16* __restrict__ outp,
                            int nheads, float scale) {
  const int idx = blockIdx.x * 256 + threadIdx.x;
  const int i  = idx & (HDc / 2 - 1);
  const int t  = idx >> 5;            // (b*S + s)*nheads + h
  const int h  = t % nheads;
  const int bs = t / nheads;
  const int s  = bs & (Sc - 1);
  const int b  = bs >> 11;
  const float c  = cosb[s * (HDc / 2) + i];
  const float sn = sinb[s * (HDc / 2) + i];
  const size_t src = (size_t)t * HDc + 2 * i;
  const float t0 = in[src], t1 = in[src + 1];
  const size_t dst = (((size_t)(b * nheads + h) * Sc + s) * HDc) + 2 * i;
  outp[dst]     = (__bf16)((t0 * c - t1 * sn) * scale);
  outp[dst + 1] = (__bf16)((t0 * sn + t1 * c) * scale);
}

// fp32 V [B,S,HK,HD] -> bf16 V^T [B,HK,HD,S]
__global__ void vtrans_kernel(const float* __restrict__ in, __bf16* __restrict__ outp) {
  const int idx = blockIdx.x * 256 + threadIdx.x;   // [b][s][hk][hd]
  const int hd = idx & 63, hk = (idx >> 6) & 3, s = (idx >> 8) & (Sc - 1), b = idx >> 19;
  outp[(((size_t)(b * HKc + hk)) * HDc + hd) * Sc + s] = (__bf16)in[idx];
}

// ---------------------------------------------------------------------------
// Sliding-window causal flash attention, transposed formulation.
//   qb: bf16 [B,H,S,HD] pre-scaled by 1/sqrt(HD)
//   kb: bf16 [B,HK,S,HD]
//   vt: bf16 [B,HK,HD,S]   (V^T so V-fragments load contiguously)
//   out: bf16 [B,S,H*HD]
// One wave per (b,h,16-query tile).  S^T = K_tile @ Q^T (two 16x16 score
// tiles per 32-key block), online softmax per query (lane pair n / n+16),
// P^T built as packed bf16x2 B-fragment via statically-indexed shuffles,
// O^T += V^T @ P^T.  All tile-level control values flow through
// readfirstlane so loops/branches are scalar; masking runs only on the
// (rare) edge blocks.
// ---------------------------------------------------------------------------
__global__ __launch_bounds__(256) void attn_kernel(
    const __bf16* __restrict__ qb, const __bf16* __restrict__ kbm,
    const __bf16* __restrict__ vtm, __bf16* __restrict__ outb) {
  const int wave = __builtin_amdgcn_readfirstlane(threadIdx.x >> 5);  // uniform
  const int lane = threadIdx.x & 31;
  const int half = lane >> 4, ln = lane & 15;
  const int tile = blockIdx.x * 8 + wave;     // B*H*(S/16) tiles (uniform)
  const int qt = tile & (Sc / 16 - 1);
  const int h  = (tile >> 7) & (Hc - 1);
  const int b  = tile >> 11;
  const int hk = h >> 2;                      // H/HK = 4
  const int q0 = qt * 16;                     // uniform
  const int qpos = q0 + ln;                   // per-lane query position

  const __bf16* qbase = qb  + (((size_t)(b * Hc  + h )) * Sc + q0) * HDc;
  const __bf16* kbase = kbm + (((size_t)(b * HKc + hk)) * Sc) * HDc;
  const __bf16* vbase = vtm + (((size_t)(b * HKc + hk)) * HDc) * Sc;

  // Q^T B-fragments for the two HD k-steps (loop invariant)
  AB16 qf[2];
  {
    const __bf16* qrow = qbase + (size_t)ln * HDc;
#pragma unroll
    for (int s2 = 0; s2 < 2; ++s2)
#pragma unroll
      for (int v2 = 0; v2 < 8; ++v2)
        qf[s2].u[v2] = *(const unsigned*)(qrow + s2 * 32 + (v2 >> 2) * 16 + half * 8 + 2 * (v2 & 3));
  }

  v8f o0 = v8zero(), o1 = v8zero(), o2 = v8zero(), o3 = v8zero();
  float rmax = -1e30f, rsum = 0.f;

  int kstart = q0 - (WINc - 1); if (kstart < 0) kstart = 0; kstart &= ~31;
  const int kend = q0 + 15;                   // uniform loop bounds

  for (int kblk = kstart; kblk <= kend; kblk += 32) {
    // ---- scores (transposed): two 16key x 16query tiles
    v8f sT0 = v8zero(), sT1 = v8zero();
#pragma unroll
    for (int s2 = 0; s2 < 2; ++s2) {
      AB16 kf0, kf1;
      const __bf16* kr0 = kbase + (size_t)(kblk + ln) * HDc + s2 * 32;
      const __bf16* kr1 = kr0 + 16 * HDc;
#pragma unroll
      for (int v2 = 0; v2 < 4; ++v2) {
        kf0.u[v2]     = *(const unsigned*)(kr0 +      half * 8 + 2 * v2);
        kf0.u[4 + v2] = *(const unsigned*)(kr0 + 16 + half * 8 + 2 * v2);
        kf1.u[v2]     = *(const unsigned*)(kr1 +      half * 8 + 2 * v2);
        kf1.u[4 + v2] = *(const unsigned*)(kr1 + 16 + half * 8 + 2 * v2);
      }
      sT0 = __builtin_amdgcn_wmma_f32_16x16x32_bf16(false, kf0.v, false, qf[s2].v,
                                                    (short)0, sT0, false, false);
      sT1 = __builtin_amdgcn_wmma_f32_16x16x32_bf16(false, kf1.v, false, qf[s2].v,
                                                    (short)0, sT1, false, false);
    }

    // edge iff any (query,key) pair in this block needs masking; scalar branch
    const bool edge = (kblk + 31 > q0) || (kblk < q0 + 16 - WINc);

    // ---- block max (masked only on edge blocks)
    unsigned ok0 = 0xffu, ok1 = 0xffu;
    float mb = -1e30f;
    if (edge) {
      ok0 = 0; ok1 = 0;
#pragma unroll
      for (int v2 = 0; v2 < 8; ++v2) {
        const int kp0 = kblk + v2 + half * 8;
        const int kp1 = kp0 + 16;
        const bool b0 = (kp0 <= qpos) && (qpos - kp0 < WINc);
        const bool b1 = (kp1 <= qpos) && (qpos - kp1 < WINc);
        const float s0v = b0 ? sT0[v2] : -1e30f;
        const float s1v = b1 ? sT1[v2] : -1e30f;
        sT0[v2] = s0v; sT1[v2] = s1v;
        ok0 |= (unsigned)b0 << v2; ok1 |= (unsigned)b1 << v2;
        mb = fmaxf(mb, fmaxf(s0v, s1v));
      }
    } else {
#pragma unroll
      for (int v2 = 0; v2 < 8; ++v2)
        mb = fmaxf(mb, fmaxf(sT0[v2], sT1[v2]));
    }
    mb = fmaxf(mb, __shfl_xor(mb, 16, 32));     // lane n <-> n+16 hold same query
    const float mnew  = fmaxf(rmax, mb);
    const float alpha = __expf(rmax - mnew);
    rmax = mnew;

    float p0a[8], p1a[8], ls = 0.f;
    if (edge) {
#pragma unroll
      for (int v2 = 0; v2 < 8; ++v2) {
        const float e0 = ((ok0 >> v2) & 1) ? __expf(sT0[v2] - mnew) : 0.f;
        const float e1 = ((ok1 >> v2) & 1) ? __expf(sT1[v2] - mnew) : 0.f;
        p0a[v2] = e0; p1a[v2] = e1; ls += e0 + e1;
      }
    } else {
#pragma unroll
      for (int v2 = 0; v2 < 8; ++v2) {          // fast path: no selects
        const float e0 = __expf(sT0[v2] - mnew);
        const float e1 = __expf(sT1[v2] - mnew);
        p0a[v2] = e0; p1a[v2] = e1; ls += e0 + e1;
      }
    }
    ls += __shfl_xor(ls, 16, 32);
    rsum = rsum * alpha + ls;

#pragma unroll
    for (int v2 = 0; v2 < 8; ++v2) {
      o0[v2] *= alpha; o1[v2] *= alpha; o2[v2] *= alpha; o3[v2] *= alpha;
    }

    // ---- P^T as B-fragment: pack bf16x2 first, then one shuffle per dword
    unsigned pk0[4], pk1[4];
#pragma unroll
    for (int j = 0; j < 4; ++j) {               // 8x v_cvt_pk_bf16_f32
      pk0[j] = pack_bf16x2(p0a[2 * j], p0a[2 * j + 1]);
      pk1[j] = pack_bf16x2(p1a[2 * j], p1a[2 * j + 1]);
    }
    AB16 pf;
#pragma unroll
    for (int v2 = 0; v2 < 8; ++v2) {
      const int j = v2 & 3;                     // source dword (keys 2j,2j+1 of tile)
      const int srcl = ln + ((v2 >= 4) ? 16 : 0);
      const unsigned a0 = (unsigned)__shfl((int)pk0[j], srcl, 32);
      const unsigned a1 = (unsigned)__shfl((int)pk1[j], srcl, 32);
      pf.u[v2] = half ? a1 : a0;                // tile select by dest half
    }

    // ---- O^T += V^T @ P^T  (4 hd tiles of 16)
#pragma unroll
    for (int t = 0; t < 4; ++t) {
      AB16 vf;
      const __bf16* vrow = vbase + (size_t)(t * 16 + ln) * Sc + kblk;
#pragma unroll
      for (int v2 = 0; v2 < 4; ++v2) {
        vf.u[v2]     = *(const unsigned*)(vrow + half * 16 +     2 * v2);
        vf.u[4 + v2] = *(const unsigned*)(vrow + half * 16 + 8 + 2 * v2);
      }
      v8f& ot = (t == 0) ? o0 : (t == 1) ? o1 : (t == 2) ? o2 : o3;
      ot = __builtin_amdgcn_wmma_f32_16x16x32_bf16(false, vf.v, false, pf.v,
                                                   (short)0, ot, false, false);
    }
  }

  const float inv = 1.f / rsum;
  const size_t obase = ((size_t)(b * Sc + q0 + ln)) * (Hc * HDc) + h * HDc;
#pragma unroll
  for (int t = 0; t < 4; ++t) {
    const v8f& ot = (t == 0) ? o0 : (t == 1) ? o1 : (t == 2) ? o2 : o3;
#pragma unroll
    for (int v2 = 0; v2 < 8; ++v2)
      outb[obase + t * 16 + v2 + half * 8] = (__bf16)(ot[v2] * inv);
  }
}

// ---------------------------------------------------------------------------
// Router: logits = fn @ router_w, softmax, top-2 -> dense gate matrix [T,E].
// One wave per token.
// ---------------------------------------------------------------------------
__global__ __launch_bounds__(256) void router_kernel(const __bf16* __restrict__ fn,
    const float* __restrict__ rwt, float* __restrict__ gmat) {
  const int token = blockIdx.x * 8 + __builtin_amdgcn_readfirstlane(threadIdx.x >> 5);
  const int lane = threadIdx.x & 31;
  const __bf16* xr = fn + (size_t)token * Dc;
  float lg[Ec];
#pragma unroll
  for (int e = 0; e < Ec; ++e) lg[e] = 0.f;
  for (int i = lane; i < Dc; i += 32) {
    const float xv = (float)xr[i];
#pragma unroll
    for (int e = 0; e < Ec; ++e) lg[e] += xv * rwt[(size_t)i * Ec + e];
  }
#pragma unroll
  for (int e = 0; e < Ec; ++e)
#pragma unroll
    for (int o = 16; o; o >>= 1) lg[e] += __shfl_xor(lg[e], o, 32);
  float m = lg[0];
#pragma unroll
  for (int e = 1; e < Ec; ++e) m = fmaxf(m, lg[e]);
  float p[Ec];
#pragma unroll
  for (int e = 0; e < Ec; ++e) p[e] = __expf(lg[e] - m);
  int i1 = 0; float v1 = p[0];
#pragma unroll
  for (int e = 1; e < Ec; ++e) if (p[e] > v1) { v1 = p[e]; i1 = e; }
  int i2 = -1; float v2 = -1.f;
#pragma unroll
  for (int e = 0; e < Ec; ++e) if (e != i1 && p[e] > v2) { v2 = p[e]; i2 = e; }
  const float dn = 1.f / (v1 + v2);
  if (lane < Ec)
    gmat[(size_t)token * Ec + lane] =
        (lane == i1) ? v1 * dn : (lane == i2) ? v2 * dn : 0.f;
}

// ---------------------------------------------------------------------------
extern "C" void kernel_launch(void* const* d_in, const int* in_sizes, int n_in,
                              void* d_out, int out_size, void* d_ws, size_t ws_size,
                              hipStream_t stream) {
  (void)in_sizes; (void)n_in; (void)out_size; (void)ws_size;
  const float* x    = (const float*)d_in[0];
  const float* fcos = (const float*)d_in[1];
  const float* fsin = (const float*)d_in[2];
  const float* anw  = (const float*)d_in[3];
  const float* wqf  = (const float*)d_in[4];
  const float* wkf  = (const float*)d_in[5];
  const float* wvf  = (const float*)d_in[6];
  const float* wof  = (const float*)d_in[7];
  const float* fnw  = (const float*)d_in[8];
  const float* rwt  = (const float*)d_in[9];
  const float* w1f  = (const float*)d_in[10];
  const float* w2f  = (const float*)d_in[11];
  float* out = (float*)d_out;                 // [B,S,D] fp32

  char* ws = (char*)d_ws;
  size_t off = 0;
  auto take = [&](size_t bytes) -> char* {
    char* p = ws + off; off += (bytes + 255) & ~(size_t)255; return p;
  };
  __bf16* hn_bf  = (__bf16*)take((size_t)Tc * Dc * 2);
  __bf16* wq_bf  = (__bf16*)take((size_t)Dc * Dc * 2);          // [N=1024,K=1024]
  __bf16* wk_bf  = (__bf16*)take((size_t)Dc * 256 * 2);         // [256,1024]
  __bf16* wv_bf  = (__bf16*)take((size_t)Dc * 256 * 2);
  __bf16* wo_bf  = (__bf16*)take((size_t)Dc * Dc * 2);
  __bf16* w1_bf  = (__bf16*)take((size_t)Ec * Dc * DHc * 2);    // per-e [DH,D]
  __bf16* w2_bf  = (__bf16*)take((size_t)Ec * DHc * Dc * 2);    // per-e [D,DH]
  float*  qf32   = (float*)take((size_t)Tc * 1024 * 4);
  float*  kf32   = (float*)take((size_t)Tc * 256 * 4);
  float*  vf32   = (float*)take((size_t)Tc * 256 * 4);
  __bf16* q_bf   = (__bf16*)take((size_t)Tc * 1024 * 2);        // [B,H,S,HD]
  __bf16* k_bf   = (__bf16*)take((size_t)Tc * 256 * 2 + 4096);  // [B,HK,S,HD]+pad
  __bf16* v_t    = (__bf16*)take((size_t)Tc * 256 * 2 + 4096);  // [B,HK,HD,S]+pad
  __bf16* attn_bf= (__bf16*)take((size_t)Tc * 1024 * 2);
  float*  h_f32  = (float*)take((size_t)Tc * Dc * 4);
  __bf16* fn_bf  = (__bf16*)take((size_t)Tc * Dc * 2);
  float*  gmat   = (float*)take((size_t)Tc * Ec * 4);
  __bf16* mid_bf = (__bf16*)take((size_t)Tc * DHc * 2);

  const dim3 blk(256);

  // --- weight conversion (fp32 [K,N] -> bf16 [N,K]) ---
  convt_kernel<<<(Dc * Dc) / 256,  blk, 0, stream>>>(wqf, wq_bf, Dc, Dc);
  convt_kernel<<<(Dc * 256) / 256, blk, 0, stream>>>(wkf, wk_bf, Dc, 256);
  convt_kernel<<<(Dc * 256) / 256, blk, 0, stream>>>(wvf, wv_bf, Dc, 256);
  convt_kernel<<<(Dc * Dc) / 256,  blk, 0, stream>>>(wof, wo_bf, Dc, Dc);
  for (int e = 0; e < Ec; ++e) {
    convt_kernel<<<(Dc * DHc) / 256, blk, 0, stream>>>(
        w1f + (size_t)e * Dc * DHc, w1_bf + (size_t)e * DHc * Dc, Dc, DHc);
    convt_kernel<<<(DHc * Dc) / 256, blk, 0, stream>>>(
        w2f + (size_t)e * DHc * Dc, w2_bf + (size_t)e * Dc * DHc, DHc, Dc);
  }

  // --- attention pre-norm ---
  rmsnorm_kernel<<<Tc, blk, 0, stream>>>(x, anw, hn_bf, 1e-5f);

  // --- QKV projections (WMMA) ---
  gemm_bf16<0><<<dim3(Tc / 128, 1024 / 64), blk, 0, stream>>>(
      hn_bf, wq_bf, qf32, nullptr, nullptr, nullptr, nullptr, 0, Tc, 1024, Dc);
  gemm_bf16<0><<<dim3(Tc / 128, 256 / 64), blk, 0, stream>>>(
      hn_bf, wk_bf, kf32, nullptr, nullptr, nullptr, nullptr, 0, Tc, 256, Dc);
  gemm_bf16<0><<<dim3(Tc / 128, 256 / 64), blk, 0, stream>>>(
      hn_bf, wv_bf, vf32, nullptr, nullptr, nullptr, nullptr, 0, Tc, 256, Dc);

  // --- RoPE + head-major bf16 layouts; fold 1/sqrt(HD) into Q ---
  rope_kernel<<<(Tc * Hc * 32) / 256, blk, 0, stream>>>(qf32, fcos, fsin, q_bf, Hc, 0.125f);
  rope_kernel<<<(Tc * HKc * 32) / 256, blk, 0, stream>>>(kf32, fcos, fsin, k_bf, HKc, 1.f);
  vtrans_kernel<<<(Tc * HKc * HDc) / 256, blk, 0, stream>>>(vf32, v_t);

  // --- sliding-window flash attention (WMMA) ---
  attn_kernel<<<(Bc * Hc * (Sc / 16)) / 8, blk, 0, stream>>>(q_bf, k_bf, v_t, attn_bf);

  // --- output projection + residual; writes h (ws) and d_out ---
  gemm_bf16<1><<<dim3(Tc / 128, Dc / 64), blk, 0, stream>>>(
      attn_bf, wo_bf, h_f32, out, nullptr, x, nullptr, 0, Tc, Dc, Dc);

  // --- FFN pre-norm + router ---
  rmsnorm_kernel<<<Tc, blk, 0, stream>>>(h_f32, fnw, fn_bf, 1e-6f);
  router_kernel<<<Tc / 8, blk, 0, stream>>>(fn_bf, rwt, gmat);

  // --- MoE: dense over experts, gated accumulate into d_out ---
  for (int e = 0; e < Ec; ++e) {
    gemm_bf16<2><<<dim3(Tc / 128, DHc / 64), blk, 0, stream>>>(
        fn_bf, w1_bf + (size_t)e * DHc * Dc, nullptr, nullptr, mid_bf,
        nullptr, nullptr, 0, Tc, DHc, Dc);
    gemm_bf16<3><<<dim3(Tc / 128, Dc / 64), blk, 0, stream>>>(
        mid_bf, w2_bf + (size_t)e * Dc * DHc, out, nullptr, nullptr,
        nullptr, gmat + e, Ec, Tc, Dc, DHc);
  }
}
